// BasicDeconvolutionBlock_7868380086734
// MI455X (gfx1250) — matrix operand order repaired and checked by reading, not compiled, not run on hardware
//
#include <hip/hip_runtime.h>

// ---------------------------------------------------------------------------
// Sparse transposed Conv3d (gather-GEMM-scatter over 27 offsets) + BN + ReLU
// for MI455X (gfx1250, wave32, WMMA).
//
// Memory-bound (~2-3 GB traffic vs 29 GFLOP), so we use bf16 WMMA with
// hi/lo split-precision (3 WMMAs per product: hi*hi + hi*lo + lo*hi) to get
// near-fp32 accuracy on the fast 16x16x32 bf16 matrix path.
// LDS rows are padded to 72 halfwords (144 B) so every WMMA fragment base is
// 16-byte aligned -> ds_load_b128 fragment fetches, minimal address VALU.
// ---------------------------------------------------------------------------

#define K_OFF   27
#define M_PAIRS 131072
#define N_VOX   524288
#define C_IN    64
#define C_OUT   64
#define BN_EPS  1e-5f

#define TM       128   // gathered rows per block
#define A_STRIDE 72    // LDS row stride in ushorts (64 data + 8 pad) = 144 B
#define B_STRIDE 72

typedef __attribute__((ext_vector_type(16))) __bf16        v16bf;
typedef __attribute__((ext_vector_type(8)))  float         v8f;
typedef __attribute__((ext_vector_type(8)))  unsigned int  v8u;

union Frag { v8u u; v16bf b; uint4 q[2]; };

__device__ __forceinline__ unsigned short f2bf(float f) {
    unsigned int u = __builtin_bit_cast(unsigned int, f);
    u += 0x7FFFu + ((u >> 16) & 1u);           // round-to-nearest-even
    return (unsigned short)(u >> 16);
}
__device__ __forceinline__ float bf2f(unsigned short h) {
    unsigned int u = ((unsigned int)h) << 16;
    return __builtin_bit_cast(float, u);
}
__device__ __forceinline__ unsigned int pack2(unsigned short lo, unsigned short hi) {
    return (unsigned int)lo | ((unsigned int)hi << 16);
}

// ---------------------------------------------------------------------------
// Kernel 1: per-offset gather -> bf16 split -> WMMA GEMM -> atomic scatter-add
// grid = (M_PAIRS/TM, K_OFF), block = 256 threads = 8 wave32
// ---------------------------------------------------------------------------
__global__ __launch_bounds__(256) void deconv_gather_wmma_scatter(
    const float* __restrict__ x, const float* __restrict__ W,
    const int* __restrict__ in_idx, const int* __restrict__ out_idx,
    float* __restrict__ out)
{
    __shared__ __align__(16) unsigned short sBhi[C_OUT * B_STRIDE]; // W[k]^T bf16 hi
    __shared__ __align__(16) unsigned short sBlo[C_OUT * B_STRIDE]; // residual lo
    __shared__ __align__(16) unsigned short sAhi[TM * A_STRIDE];    // gathered rows hi
    __shared__ __align__(16) unsigned short sAlo[TM * A_STRIDE];    // residual lo

    const int k   = blockIdx.y;
    const int m0  = blockIdx.x * TM;
    const int tid = threadIdx.x;

    // ---- stage W[k] (C_IN x C_OUT row-major) into LDS transposed (co-major)
    const float* Wk = W + (size_t)k * C_IN * C_OUT;
    for (int i = tid; i < C_IN * C_OUT; i += 256) {
        int ci = i >> 6, co = i & 63;
        float w = Wk[i];
        unsigned short hi = f2bf(w);
        sBhi[co * B_STRIDE + ci] = hi;
        sBlo[co * B_STRIDE + ci] = f2bf(w - bf2f(hi));
    }

    // ---- gather TM rows of x (float4 loads), split into bf16 hi/lo, packed b64 stores
    const int* inI = in_idx + (size_t)k * M_PAIRS + m0;
    for (int i = tid; i < TM * (C_IN / 4); i += 256) {
        int r = i >> 4;               // row in tile
        int c = (i & 15) << 2;        // channel base (multiple of 4)
        int g = inI[r];
        float4 v = *reinterpret_cast<const float4*>(x + (size_t)g * C_IN + c);
        unsigned short h0 = f2bf(v.x), h1 = f2bf(v.y), h2 = f2bf(v.z), h3 = f2bf(v.w);
        uint2 hi2, lo2;
        hi2.x = pack2(h0, h1);
        hi2.y = pack2(h2, h3);
        lo2.x = pack2(f2bf(v.x - bf2f(h0)), f2bf(v.y - bf2f(h1)));
        lo2.y = pack2(f2bf(v.z - bf2f(h2)), f2bf(v.w - bf2f(h3)));
        *reinterpret_cast<uint2*>(sAhi + r * A_STRIDE + c) = hi2;   // 8B aligned
        *reinterpret_cast<uint2*>(sAlo + r * A_STRIDE + c) = lo2;
    }
    __syncthreads();

    const int wave   = tid >> 5;
    const int lane   = tid & 31;
    const int lh     = lane >> 4;     // lane-half select
    const int lm     = lane & 15;
    const int mstrip = wave * 16;     // this wave owns rows [mstrip, mstrip+16)

    v8f acc[4] = {};                  // 16x64 f32 strip = 4 tiles of 16x16

#pragma unroll
    for (int kc = 0; kc < 2; ++kc) {  // K = 64 = 2 chunks of 32
        // A fragment per ISA 16-bit 16x32 layout:
        //   lane<16: K = 2j,2j+1 (v0-3), 16+2j.. (v4-7); lane>=16: +8
        // Fragment = dwords [4*lh .. 4*lh+3] and [8+4*lh .. 11+4*lh] of the row
        // chunk -> two 16B-aligned ds_load_b128.
        Frag ahi, alo;
        const uint4* aH =
            (const uint4*)(sAhi + (mstrip + lm) * A_STRIDE + kc * 32);
        const uint4* aL =
            (const uint4*)(sAlo + (mstrip + lm) * A_STRIDE + kc * 32);
        ahi.q[0] = aH[lh];     ahi.q[1] = aH[2 + lh];
        alo.q[0] = aL[lh];     alo.q[1] = aL[2 + lh];

#pragma unroll
        for (int t = 0; t < 4; ++t) { // 4 column tiles of 16
            // B fragment (32x16), same K order as A: lane<16 -> K=0..7,16..23,
            // lane>=16 -> K=8..15,24..31, at N = lane%16 (N-major LDS layout).
            // two 16B-aligned ds_load_b128.
            Frag bhi, blo;
            const uint4* bH =
                (const uint4*)(sBhi + (t * 16 + lm) * B_STRIDE);
            const uint4* bL =
                (const uint4*)(sBlo + (t * 16 + lm) * B_STRIDE);
            const int qi = kc * 4 + lh;
            bhi.q[0] = bH[qi];  bhi.q[1] = bH[qi + 2];
            blo.q[0] = bL[qi];  blo.q[1] = bL[qi + 2];

            // split-precision f32 product: hi*hi + hi*lo + lo*hi
            acc[t] = __builtin_amdgcn_wmma_f32_16x16x32_bf16(
                false, ahi.b, false, bhi.b, (short)0, acc[t], false, false);
            acc[t] = __builtin_amdgcn_wmma_f32_16x16x32_bf16(
                false, ahi.b, false, blo.b, (short)0, acc[t], false, false);
            acc[t] = __builtin_amdgcn_wmma_f32_16x16x32_bf16(
                false, alo.b, false, bhi.b, (short)0, acc[t], false, false);
        }
    }

    // ---- scatter-add: C/D layout => lane holds N=lm, M=(lh? j+8 : j)
    const int* outI = out_idx + (size_t)k * M_PAIRS + m0 + mstrip;
#pragma unroll
    for (int j = 0; j < 8; ++j) {
        const int m   = lh ? (j + 8) : j;
        const int row = outI[m];
        float* base = out + (size_t)row * C_OUT + lm;
        atomicAdd(base +  0, acc[0][j]);
        atomicAdd(base + 16, acc[1][j]);
        atomicAdd(base + 32, acc[2][j]);
        atomicAdd(base + 48, acc[3][j]);
    }
}

// ---------------------------------------------------------------------------
// Kernel 2: per-channel sum / sum-of-squares over N_VOX rows
// grid = N_VOX/512, block = 256 (4 row-groups x 64 channels)
// ---------------------------------------------------------------------------
__global__ __launch_bounds__(256) void bn_stats(
    const float* __restrict__ out, float* __restrict__ stats)
{
    const int tid = threadIdx.x;
    const int c   = tid & 63;
    const int rg  = tid >> 6;             // 0..3
    const int r0  = blockIdx.x * 512;

    float s = 0.f, ss = 0.f;
    for (int r = rg; r < 512; r += 4) {
        float v = out[(size_t)(r0 + r) * C_OUT + c];
        s  += v;
        ss += v * v;
    }
    __shared__ float ls[4][C_OUT], lss[4][C_OUT];
    ls[rg][c]  = s;
    lss[rg][c] = ss;
    __syncthreads();
    if (rg == 0) {
        s  = ls[0][c]  + ls[1][c]  + ls[2][c]  + ls[3][c];
        ss = lss[0][c] + lss[1][c] + lss[2][c] + lss[3][c];
        atomicAdd(&stats[c], s);
        atomicAdd(&stats[C_OUT + c], ss);
    }
}

// ---------------------------------------------------------------------------
// Kernel 3: BN (batch stats) + ReLU, in place on d_out, float4 per thread
// ---------------------------------------------------------------------------
__global__ __launch_bounds__(256) void bn_apply(
    float* __restrict__ out, const float* __restrict__ stats,
    const float* __restrict__ gamma, const float* __restrict__ beta)
{
    __shared__ float sScale[C_OUT], sShift[C_OUT];
    const int tid = threadIdx.x;
    if (tid < C_OUT) {
        const float invN = 1.0f / (float)N_VOX;
        float mean = stats[tid] * invN;
        float var  = stats[C_OUT + tid] * invN - mean * mean;  // biased, as jnp.var
        float sc   = rsqrtf(var + BN_EPS) * gamma[tid];
        sScale[tid] = sc;
        sShift[tid] = beta[tid] - mean * sc;
    }
    __syncthreads();

    size_t i = (size_t)blockIdx.x * 256 + tid;   // float4 index
    float4* p = reinterpret_cast<float4*>(out);
    float4 v = p[i];
    int c = (int)((i & 15) << 2);                // 4 consecutive channels
    v.x = fmaxf(fmaf(v.x, sScale[c + 0], sShift[c + 0]), 0.f);
    v.y = fmaxf(fmaf(v.y, sScale[c + 1], sShift[c + 1]), 0.f);
    v.z = fmaxf(fmaf(v.z, sScale[c + 2], sShift[c + 2]), 0.f);
    v.w = fmaxf(fmaf(v.w, sScale[c + 3], sShift[c + 3]), 0.f);
    p[i] = v;
}

// ---------------------------------------------------------------------------
extern "C" void kernel_launch(void* const* d_in, const int* in_sizes, int n_in,
                              void* d_out, int out_size, void* d_ws, size_t ws_size,
                              hipStream_t stream)
{
    (void)in_sizes; (void)n_in; (void)out_size; (void)ws_size;

    const float* x       = (const float*)d_in[0];
    const float* W       = (const float*)d_in[1];
    const float* gamma   = (const float*)d_in[2];
    const float* beta    = (const float*)d_in[3];
    const int*   in_idx  = (const int*)d_in[4];
    const int*   out_idx = (const int*)d_in[5];
    float*       out     = (float*)d_out;
    float*       stats   = (float*)d_ws;     // [0..63]=sum, [64..127]=sumsq

    // zero accumulator + stats (stream-ordered; graph-capture safe)
    hipMemsetAsync(out,   0, (size_t)N_VOX * C_OUT * sizeof(float), stream);
    hipMemsetAsync(stats, 0, 2 * C_OUT * sizeof(float), stream);

    dim3 grid(M_PAIRS / TM, K_OFF);          // 1024 x 27 blocks
    deconv_gather_wmma_scatter<<<grid, 256, 0, stream>>>(x, W, in_idx, out_idx, out);

    bn_stats<<<N_VOX / 512, 256, 0, stream>>>(out, stats);

    bn_apply<<<(N_VOX * C_OUT / 4) / 256, 256, 0, stream>>>(out, stats, gamma, beta);
}
